// PolarizedSelfAttention_79809082294689
// MI455X (gfx1250) — matrix operand order repaired and checked by reading, not compiled, not run on hardware
//
#include <hip/hip_runtime.h>
#include <math.h>

#define B_   32
#define H_   128
#define W_   96
#define C_   128
#define CH_  64
#define HW_  (H_ * W_)            // 12288
#define TILES_PER_B 32
#define POS_PER_BLOCK (HW_ / TILES_PER_B)   // 384
#define POS_PER_WAVE  (POS_PER_BLOCK / 8)   // 48

typedef __attribute__((ext_vector_type(16))) _Float16 v16h;
typedef __attribute__((ext_vector_type(8)))  _Float16 v8h;
typedef __attribute__((ext_vector_type(8)))  float    v8f;

// ---------------- WMMA fragment helpers (wave32, ISA 7.12.2 layouts) ----------------
// A: 16x32 f16, A row-major in LDS (ld = lda halves).
// lane l: row m = l&15; K-half kh = 8*(l>>4). Halves needed: [k0+kh, k0+kh+7] and
// [k0+16+kh, k0+16+kh+7] -> two aligned 16B LDS loads.
__device__ __forceinline__ v16h load_A(const _Float16* A, int lda, int row0, int k0) {
  const int l  = threadIdx.x & 31;
  const int m  = row0 + (l & 15);
  const int kh = (l >> 4) << 3;                 // 0 or 8
  const _Float16* p = A + m * lda + k0 + kh;
  v8h lo = *reinterpret_cast<const v8h*>(p);
  v8h hi = *reinterpret_cast<const v8h*>(p + 16);
  return __builtin_shufflevector(lo, hi, 0, 1, 2, 3, 4, 5, 6, 7,
                                         8, 9, 10, 11, 12, 13, 14, 15);
}
// B: 32x16 f16, staged TRANSPOSED in LDS: Bt[n][k] (ld = ldk halves).
// lane l: col n = l&15; K-half kh = 8*(l>>4). Halves: [k0+kh, k0+kh+7] and
// [k0+16+kh, k0+16+kh+7] -> two aligned 16B LDS loads.
__device__ __forceinline__ v16h load_Bt(const _Float16* Bt, int ldk, int k0, int col0) {
  const int l  = threadIdx.x & 31;
  const int n  = col0 + (l & 15);
  const int kh = (l >> 4) << 3;                 // 0 or 8
  const _Float16* p = Bt + n * ldk + k0 + kh;
  v8h lo = *reinterpret_cast<const v8h*>(p);
  v8h hi = *reinterpret_cast<const v8h*>(p + 16);
  return __builtin_shufflevector(lo, hi, 0, 1, 2, 3, 4, 5, 6, 7,
                                         8, 9, 10, 11, 12, 13, 14, 15);
}
// C/D: 16x16 f32. lane l: col n = l&15; VGPR r: row m = r + 8*(l>>4).
__device__ __forceinline__ void store_D_f32(float* Dm, int ldd, int row0, int col0, v8f d) {
  const int l  = threadIdx.x & 31;
  const int n  = col0 + (l & 15);
  const int mh = (l >> 4) << 3;
#pragma unroll
  for (int r = 0; r < 8; ++r) Dm[(row0 + mh + r) * ldd + n] = d[r];
}
__device__ __forceinline__ void store_D_f16(_Float16* Dm, int ldd, int row0, int col0, v8f d) {
  const int l  = threadIdx.x & 31;
  const int n  = col0 + (l & 15);
  const int mh = (l >> 4) << 3;
#pragma unroll
  for (int r = 0; r < 8; ++r) Dm[(row0 + mh + r) * ldd + n] = (_Float16)d[r];
}

__device__ __forceinline__ float wave_sum(float v) {
#pragma unroll
  for (int off = 16; off > 0; off >>= 1) v += __shfl_xor(v, off);
  return v;
}
__device__ __forceinline__ float wave_max(float v) {
#pragma unroll
  for (int off = 16; off > 0; off >>= 1) v = fmaxf(v, __shfl_xor(v, off));
  return v;
}

// ---------------- k0: zero accumulators ----------------
__global__ void k0_zero(float* __restrict__ p, int n) {
  int i = blockIdx.x * blockDim.x + threadIdx.x;
  if (i < n) p[i] = 0.0f;
}

// ---------------- k1: q_raw = x . wq_ch ; xsum = sum_n x ----------------
__global__ __launch_bounds__(256) void k1_qraw_xsum(
    const float* __restrict__ x, const float* __restrict__ wq_ch,
    float* __restrict__ q_raw, float* __restrict__ xsum) {
  const int b    = blockIdx.x / TILES_PER_B;
  const int tile = blockIdx.x % TILES_PER_B;
  const int lane = threadIdx.x & 31;
  const int wave = threadIdx.x >> 5;
  __shared__ float sXs[C_];
  if (threadIdx.x < C_) sXs[threadIdx.x] = 0.0f;
  __syncthreads();

  const float4* xb = reinterpret_cast<const float4*>(x) + (size_t)b * HW_ * (C_ / 4);
  const float4  wq = reinterpret_cast<const float4*>(wq_ch)[lane];
  float sx = 0.f, sy = 0.f, sz = 0.f, sw = 0.f;
  const int n0 = tile * POS_PER_BLOCK + wave * POS_PER_WAVE;
  for (int p = 0; p < POS_PER_WAVE; ++p) {
    const int n = n0 + p;
    const float4 xv = xb[(size_t)n * (C_ / 4) + lane];
    sx += xv.x; sy += xv.y; sz += xv.z; sw += xv.w;
    float d = xv.x * wq.x + xv.y * wq.y + xv.z * wq.z + xv.w * wq.w;
    d = wave_sum(d);
    if (lane == 0) q_raw[b * HW_ + n] = d;
  }
  atomicAdd(&sXs[lane * 4 + 0], sx);
  atomicAdd(&sXs[lane * 4 + 1], sy);
  atomicAdd(&sXs[lane * 4 + 2], sz);
  atomicAdd(&sXs[lane * 4 + 3], sw);
  __syncthreads();
  if (threadIdx.x < C_) atomicAdd(&xsum[b * C_ + threadIdx.x], sXs[threadIdx.x]);
}

// ---------------- k2: per-batch softmax stats over HW ----------------
__global__ __launch_bounds__(256) void k2_stats(
    const float* __restrict__ q_raw, float* __restrict__ mx, float* __restrict__ invZ) {
  const int b = blockIdx.x, tid = threadIdx.x;
  __shared__ float red[256];
  float m = -3.0e38f;
  for (int i = tid; i < HW_; i += 256) m = fmaxf(m, q_raw[b * HW_ + i]);
  red[tid] = m; __syncthreads();
  for (int s = 128; s > 0; s >>= 1) { if (tid < s) red[tid] = fmaxf(red[tid], red[tid + s]); __syncthreads(); }
  m = red[0]; __syncthreads();
  float sum = 0.f;
  for (int i = tid; i < HW_; i += 256) sum += __expf(q_raw[b * HW_ + i] - m);
  red[tid] = sum; __syncthreads();
  for (int s = 128; s > 0; s >>= 1) { if (tid < s) red[tid] += red[tid + s]; __syncthreads(); }
  if (tid == 0) { mx[b] = m; invZ[b] = 1.0f / red[0]; }
}

// ---------------- k3: xq = sum_n softmax(q)_n * x_n ----------------
__global__ __launch_bounds__(256) void k3_xq(
    const float* __restrict__ x, const float* __restrict__ q_raw,
    const float* __restrict__ mx, const float* __restrict__ invZ,
    float* __restrict__ xq) {
  const int b    = blockIdx.x / TILES_PER_B;
  const int tile = blockIdx.x % TILES_PER_B;
  const int lane = threadIdx.x & 31;
  const int wave = threadIdx.x >> 5;
  __shared__ float sXq[C_];
  if (threadIdx.x < C_) sXq[threadIdx.x] = 0.0f;
  __syncthreads();

  const float m  = mx[b];
  const float iz = invZ[b];
  const float4* xb = reinterpret_cast<const float4*>(x) + (size_t)b * HW_ * (C_ / 4);
  float ax = 0.f, ay = 0.f, az = 0.f, aw = 0.f;
  const int n0 = tile * POS_PER_BLOCK + wave * POS_PER_WAVE;
  for (int p = 0; p < POS_PER_WAVE; ++p) {
    const int n = n0 + p;
    const float pr = __expf(q_raw[b * HW_ + n] - m) * iz;   // wave-uniform broadcast load
    const float4 xv = xb[(size_t)n * (C_ / 4) + lane];
    ax += pr * xv.x; ay += pr * xv.y; az += pr * xv.z; aw += pr * xv.w;
  }
  atomicAdd(&sXq[lane * 4 + 0], ax);
  atomicAdd(&sXq[lane * 4 + 1], ay);
  atomicAdd(&sXq[lane * 4 + 2], az);
  atomicAdd(&sXq[lane * 4 + 3], aw);
  __syncthreads();
  if (threadIdx.x < C_) atomicAdd(&xq[b * C_ + threadIdx.x], sXq[threadIdx.x]);
}

// ---------------- k4: four small WMMA GEMMs + LN/sigmoid/softmax ----------------
__global__ __launch_bounds__(256) void k4_small(
    const float* __restrict__ wv_ch, const float* __restrict__ w_ch,
    const float* __restrict__ gamma, const float* __restrict__ beta,
    const float* __restrict__ wq_sp, const float* __restrict__ wv_sp,
    const float* __restrict__ xq,    const float* __restrict__ xsum,
    float* __restrict__ mask_ch, float* __restrict__ w_eff) {
  __shared__ __align__(16) _Float16 sWt[C_ * CH_];   // weight, TRANSPOSED: [n][k]
  __shared__ __align__(16) _Float16 sA [B_ * C_];    // activations (xq / xsum)
  __shared__ __align__(16) _Float16 sA2[B_ * CH_];   // ctx_half / q_sp
  __shared__ float sF[B_ * C_];                      // f32 GEMM results
  const int tid  = threadIdx.x;
  const int lane = tid & 31;
  const int wave = tid >> 5;

  // stage wv_ch^T: sWt[n*128 + k] = wv_ch[k*64 + n]; and xq [B,C]
  for (int i = tid; i < C_ * CH_; i += 256) {
    const int n = i >> 7, k = i & 127;
    sWt[i] = (_Float16)wv_ch[k * CH_ + n];
  }
  for (int i = tid; i < B_ * C_; i += 256) sA[i] = (_Float16)xq[i];
  __syncthreads();

  // GEMM1: ctx_half[32,64] = xq @ wv_ch   (8 tiles, 1 per wave)
  {
    const int mi = wave >> 2, ni = wave & 3;
    v8f acc = {};
#pragma unroll
    for (int kk = 0; kk < C_; kk += 32) {
      v16h a = load_A(sA, C_, mi * 16, kk);
      v16h b = load_Bt(sWt, C_, kk, ni * 16);
      acc = __builtin_amdgcn_wmma_f32_16x16x32_f16(false, a, false, b, (short)0, acc, false, false);
    }
    store_D_f16(sA2, CH_, mi * 16, ni * 16, acc);
  }
  __syncthreads();

  // stage w_ch^T: sWt[n*64 + k] = w_ch[k*128 + n]
  for (int i = tid; i < CH_ * C_; i += 256) {
    const int n = i >> 6, k = i & 63;
    sWt[i] = (_Float16)w_ch[k * C_ + n];
  }
  __syncthreads();

  // GEMM2: ctx[32,128] = ctx_half @ w_ch  (16 tiles, 2 per wave)
#pragma unroll
  for (int j = 0; j < 2; ++j) {
    const int t = wave * 2 + j;
    const int mi = t & 1, ni = t >> 1;
    v8f acc = {};
#pragma unroll
    for (int kk = 0; kk < CH_; kk += 32) {
      v16h a = load_A(sA2, CH_, mi * 16, kk);
      v16h b = load_Bt(sWt, CH_, kk, ni * 16);
      acc = __builtin_amdgcn_wmma_f32_16x16x32_f16(false, a, false, b, (short)0, acc, false, false);
    }
    store_D_f32(sF, C_, mi * 16, ni * 16, acc);
  }
  __syncthreads();

  // LayerNorm over C + affine + sigmoid -> mask_ch (wave handles 4 rows)
  for (int j = 0; j < 4; ++j) {
    const int r = wave * 4 + j;
    const float v0 = sF[r * C_ + lane +  0];
    const float v1 = sF[r * C_ + lane + 32];
    const float v2 = sF[r * C_ + lane + 64];
    const float v3 = sF[r * C_ + lane + 96];
    const float mu = wave_sum(v0 + v1 + v2 + v3) * (1.0f / C_);
    const float d0 = v0 - mu, d1 = v1 - mu, d2 = v2 - mu, d3 = v3 - mu;
    const float var  = wave_sum(d0 * d0 + d1 * d1 + d2 * d2 + d3 * d3) * (1.0f / C_);
    const float rstd = rsqrtf(var + 1e-5f);
    const float g0 = gamma[lane], g1 = gamma[lane + 32], g2 = gamma[lane + 64], g3 = gamma[lane + 96];
    const float b0 = beta [lane], b1 = beta [lane + 32], b2 = beta [lane + 64], b3 = beta [lane + 96];
    mask_ch[r * C_ + lane +  0] = 1.0f / (1.0f + __expf(-(d0 * rstd * g0 + b0)));
    mask_ch[r * C_ + lane + 32] = 1.0f / (1.0f + __expf(-(d1 * rstd * g1 + b1)));
    mask_ch[r * C_ + lane + 64] = 1.0f / (1.0f + __expf(-(d2 * rstd * g2 + b2)));
    mask_ch[r * C_ + lane + 96] = 1.0f / (1.0f + __expf(-(d3 * rstd * g3 + b3)));
  }
  __syncthreads();

  // stage wq_sp^T and xsum/HW
  const float inv_hw = 1.0f / (float)HW_;
  for (int i = tid; i < C_ * CH_; i += 256) {
    const int n = i >> 7, k = i & 127;
    sWt[i] = (_Float16)wq_sp[k * CH_ + n];
  }
  for (int i = tid; i < B_ * C_; i += 256) sA[i] = (_Float16)(xsum[i] * inv_hw);
  __syncthreads();

  // GEMM3: q_sp_raw[32,64] = (xsum/HW) @ wq_sp
  {
    const int mi = wave >> 2, ni = wave & 3;
    v8f acc = {};
#pragma unroll
    for (int kk = 0; kk < C_; kk += 32) {
      v16h a = load_A(sA, C_, mi * 16, kk);
      v16h b = load_Bt(sWt, C_, kk, ni * 16);
      acc = __builtin_amdgcn_wmma_f32_16x16x32_f16(false, a, false, b, (short)0, acc, false, false);
    }
    store_D_f32(sF, CH_, mi * 16, ni * 16, acc);
  }
  __syncthreads();

  // softmax over 64 per row -> q_sp (f16 in sA2)
  for (int j = 0; j < 4; ++j) {
    const int r = wave * 4 + j;
    const float v0 = sF[r * CH_ + lane];
    const float v1 = sF[r * CH_ + lane + 32];
    const float m  = wave_max(fmaxf(v0, v1));
    const float e0 = __expf(v0 - m), e1 = __expf(v1 - m);
    const float is = 1.0f / wave_sum(e0 + e1);
    sA2[r * CH_ + lane]      = (_Float16)(e0 * is);
    sA2[r * CH_ + lane + 32] = (_Float16)(e1 * is);
  }
  // GEMM4's B is wv_sp^T [64=d rows][128=c cols]; its transpose is wv_sp itself:
  // sWt[c*64 + d] = wv_sp[c*64 + d]  (contiguous copy)
  for (int i = tid; i < C_ * CH_; i += 256) sWt[i] = (_Float16)wv_sp[i];
  __syncthreads();

  // GEMM4: w_eff[32,128] = q_sp @ wv_sp^T  -> global
#pragma unroll
  for (int j = 0; j < 2; ++j) {
    const int t = wave * 2 + j;
    const int mi = t & 1, ni = t >> 1;
    v8f acc = {};
#pragma unroll
    for (int kk = 0; kk < CH_; kk += 32) {
      v16h a = load_A(sA2, CH_, mi * 16, kk);
      v16h b = load_Bt(sWt, CH_, kk, ni * 16);
      acc = __builtin_amdgcn_wmma_f32_16x16x32_f16(false, a, false, b, (short)0, acc, false, false);
    }
    store_D_f32(w_eff, C_, mi * 16, ni * 16, acc);
  }
}

// ---------------- k5: out = x*(mask_ch + sigmoid(x . w_eff[b])) ----------------
__global__ __launch_bounds__(256) void k5_out(
    const float* __restrict__ x, const float* __restrict__ mask_ch,
    const float* __restrict__ w_eff, float* __restrict__ out) {
  const int b    = blockIdx.x / TILES_PER_B;
  const int tile = blockIdx.x % TILES_PER_B;
  const int lane = threadIdx.x & 31;
  const int wave = threadIdx.x >> 5;
  const float4 we = reinterpret_cast<const float4*>(w_eff)  [b * (C_ / 4) + lane];
  const float4 mk = reinterpret_cast<const float4*>(mask_ch)[b * (C_ / 4) + lane];
  const float4* xb = reinterpret_cast<const float4*>(x) + (size_t)b * HW_ * (C_ / 4);
  float4* ob = reinterpret_cast<float4*>(out) + (size_t)b * HW_ * (C_ / 4);
  const int n0 = tile * POS_PER_BLOCK + wave * POS_PER_WAVE;
  for (int p = 0; p < POS_PER_WAVE; ++p) {
    const int n = n0 + p;
    const float4 xv = xb[(size_t)n * (C_ / 4) + lane];
    float d = xv.x * we.x + xv.y * we.y + xv.z * we.z + xv.w * we.w;
    d = wave_sum(d);                              // every lane gets the dot
    const float sg = 1.0f / (1.0f + __expf(-d));
    float4 o;
    o.x = xv.x * (mk.x + sg);
    o.y = xv.y * (mk.y + sg);
    o.z = xv.z * (mk.z + sg);
    o.w = xv.w * (mk.w + sg);
    ob[(size_t)n * (C_ / 4) + lane] = o;
  }
}

extern "C" void kernel_launch(void* const* d_in, const int* in_sizes, int n_in,
                              void* d_out, int out_size, void* d_ws, size_t ws_size,
                              hipStream_t stream) {
  (void)in_sizes; (void)n_in; (void)out_size; (void)ws_size;
  const float* x     = (const float*)d_in[0];
  const float* wq_ch = (const float*)d_in[1];
  const float* wv_ch = (const float*)d_in[2];
  const float* w_ch  = (const float*)d_in[3];
  const float* gamma = (const float*)d_in[4];
  const float* beta  = (const float*)d_in[5];
  const float* wq_sp = (const float*)d_in[6];
  const float* wv_sp = (const float*)d_in[7];
  float* out = (float*)d_out;

  float* ws    = (float*)d_ws;
  float* q_raw = ws;                               // B*HW
  float* xsum  = q_raw + (size_t)B_ * HW_;         // B*C (xsum and xq adjacent -> one zero pass)
  float* xq    = xsum + B_ * C_;                   // B*C
  float* mx    = xq + B_ * C_;                     // B
  float* invZ  = mx + B_;                          // B
  float* mask  = invZ + B_;                        // B*C
  float* weff  = mask + B_ * C_;                   // B*C

  k0_zero<<<dim3((2 * B_ * C_ + 255) / 256), 256, 0, stream>>>(xsum, 2 * B_ * C_);
  k1_qraw_xsum<<<B_ * TILES_PER_B, 256, 0, stream>>>(x, wq_ch, q_raw, xsum);
  k2_stats<<<B_, 256, 0, stream>>>(q_raw, mx, invZ);
  k3_xq<<<B_ * TILES_PER_B, 256, 0, stream>>>(x, q_raw, mx, invZ, xq);
  k4_small<<<1, 256, 0, stream>>>(wv_ch, w_ch, gamma, beta, wq_sp, wv_sp, xq, xsum, mask, weff);
  k5_out<<<B_ * TILES_PER_B, 256, 0, stream>>>(x, mask, weff, out);
}
